// FFSNN_general_12120397709828
// MI455X (gfx1250) — matrix operand order repaired and checked by reading, not compiled, not run on hardware
//
#include <hip/hip_runtime.h>

// ---------------------------------------------------------------------------
// FFSNN on MI455X (gfx1250): persistent kernel, state in WMMA accumulators,
// LDS-resident weights, async global->LDS spike staging.
//
// Per step:   L1 elementwise -> store s1(f16) -> grid barrier
//             async-stage s1 tile to LDS -> L2 WMMA GEMM (A,B from LDS)
//             -> store s2(f16) -> grid barrier
//             async-stage s2 tile to LDS -> L3 WMMA GEMM (A,B from LDS)
// Epilogue:   outputs = (ss3 @ W4^T)/T + b4   (GEMM hoisted out of time loop)
// ---------------------------------------------------------------------------

typedef __attribute__((ext_vector_type(16))) _Float16 v16h;
typedef __attribute__((ext_vector_type(8)))  _Float16 v8h;
typedef __attribute__((ext_vector_type(8)))  float    v8f;

#define T_STEPS 784
#define NB      128
#define HD      512
#define OUTD    10
#define DECAYC  0.2f
#define THRESH  0.5f

#define BLOCKS  32
#define TPB     256
#define WPB     8          // waves per block (wave32)

#define LDS_STRIDE 520     // 512 + 8 halves pad: 1040B row stride, bank-safe
#define LDS_W2_OFF 0                         // 128*520 halves
#define LDS_W3_OFF (128 * LDS_STRIDE)        // 128*520 halves
#define LDS_AT_OFF (256 * LDS_STRIDE)        // 16*520 halves (s1/s2 staging)
#define SMEM_HALves (256 * LDS_STRIDE + 16 * LDS_STRIDE)
#define SMEM_BYTES (SMEM_HALves * 2)         // 282880 B < 320KB WGP LDS

// ---- workspace layout (byte offsets, all 256B-aligned) ----
#define WS_W2H   0                         // 512*512 f16 = 524288
#define WS_W3H   (WS_W2H + 524288)         // 524288
#define WS_W4H   (WS_W3H + 524288)         // 16*512 f16 (rows 10..15 zero)
#define WS_S1    (WS_W4H + 16384)          // 128*512 f16 = 131072
#define WS_S2    (WS_S1 + 131072)          // 131072
#define WS_SS3H  (WS_S2 + 131072)          // 131072
#define WS_CNT   (WS_SS3H + 131072)        // u32 barrier counter

// ---------------------------------------------------------------------------
__global__ void ffsnn_init_kernel(const float* __restrict__ W2,
                                  const float* __restrict__ W3,
                                  const float* __restrict__ W4,
                                  _Float16* w2h, _Float16* w3h, _Float16* w4h,
                                  unsigned* cnt, float* layer_fr) {
  int i = blockIdx.x * blockDim.x + threadIdx.x;
  int stride = gridDim.x * blockDim.x;
  for (int j = i; j < HD * HD; j += stride) {
    w2h[j] = (_Float16)W2[j];
    w3h[j] = (_Float16)W3[j];
  }
  for (int j = i; j < 16 * HD; j += stride) {
    int r = j >> 9, k = j & (HD - 1);
    w4h[j] = (r < OUTD) ? (_Float16)W4[r * HD + k] : (_Float16)0.0f;
  }
  if (i == 0) *cnt = 0u;
  if (i < 3) layer_fr[i] = 0.0f;   // re-zeroed every launch (atomic targets)
}

// ---------------------------------------------------------------------------
// Generic fragment builders (source may be global or LDS; forceinline lets
// clang's addrspace inference emit global_load_b128 / ds_load_b128 as fits).
// A: 16x32 f16 tile. Lane l: row m=l&15; half=l>>4. VGPR0-3 = K[half*8..+7],
// VGPR4-7 = K[16+half*8..+7]  (ISA 05_wmma 16-bit A layout).
__device__ __forceinline__ v16h load_a_frag(const _Float16* src, int ld,
                                            int lane) {
  int m = lane & 15, half = lane >> 4;
  const _Float16* p = src + m * ld + half * 8;
  v8h c0 = *(const v8h*)p;
  v8h c1 = *(const v8h*)(p + 16);
  v16h a;
#pragma unroll
  for (int i = 0; i < 8; ++i) { a[i] = c0[i]; a[8 + i] = c1[i]; }
  return a;
}

// B: 32x16, B[k][n] = W[n][k] with W row-major (ld stride).
// Lane l: col n=l&15; half=l>>4: elems 0-7 = K[half*8..+7], 8-15 = K[16+half*8..+7].
__device__ __forceinline__ v16h load_b_frag(const _Float16* w, int ld,
                                            int lane) {
  int n = lane & 15, half = lane >> 4;
  const _Float16* p = w + n * ld + half * 8;
  v8h c0 = *(const v8h*)p;
  v8h c1 = *(const v8h*)(p + 16);
  v16h b;
#pragma unroll
  for (int i = 0; i < 8; ++i) { b[i] = c0[i]; b[8 + i] = c1[i]; }
  return b;
}

// 16x16 tile, K=512: 16 chained v_wmma_f32_16x16x32_f16.
__device__ __forceinline__ v8f gemm512(const _Float16* A, int lda,
                                       const _Float16* W, int ldw, int lane) {
  v8f acc = {};
#pragma unroll 4
  for (int k = 0; k < HD; k += 32) {
    v16h a = load_a_frag(A + k, lda, lane);
    v16h b = load_b_frag(W + k, ldw, lane);
    acc = __builtin_amdgcn_wmma_f32_16x16x32_f16(false, a, false, b,
                                                 (short)0, acc, false, false);
  }
  return acc;
}

// Stage a 16x512 f16 tile (row-major, ld=512) into LDS (stride LDS_STRIDE)
// with CDNA5 async global->LDS copies (tracked by ASYNCcnt).
__device__ __forceinline__ void stage_tile_async(const _Float16* __restrict__ g,
                                                 _Float16* ldsA, int tid) {
#pragma unroll
  for (int c = 0; c < 4; ++c) {
    int chunk = tid * 4 + c;            // 1024 x 16B chunks
    int row = chunk >> 6;               // 0..15
    int k   = (chunk & 63) * 8;         // 0..504
    unsigned       dst = (unsigned)(uintptr_t)(ldsA + row * LDS_STRIDE + k);
    unsigned long long src = (unsigned long long)(uintptr_t)(g + row * HD + k);
    asm volatile("global_load_async_to_lds_b128 %0, %1, off"
                 :: "v"(dst), "v"(src) : "memory");
  }
  asm volatile("s_wait_asynccnt 0x0" ::: "memory");
  __syncthreads();                      // all waves' LDS writes visible
}

// Monotonic-target grid barrier (all BLOCKS blocks co-resident).
__device__ __forceinline__ void grid_sync(unsigned* cnt, unsigned target) {
  __threadfence();
  __syncthreads();
  if (threadIdx.x == 0) {
    atomicAdd(cnt, 1u);
    while (__hip_atomic_load(cnt, __ATOMIC_RELAXED,
                             __HIP_MEMORY_SCOPE_AGENT) < target) {
      __builtin_amdgcn_s_sleep(1);
    }
  }
  __syncthreads();
  __threadfence();
}

// ---------------------------------------------------------------------------
__global__ void __launch_bounds__(TPB, 1)
ffsnn_main_kernel(const float* __restrict__ x,   const float* __restrict__ W1,
                  const float* __restrict__ b1,  const float* __restrict__ b2,
                  const float* __restrict__ b3,  const float* __restrict__ b4,
                  const float* __restrict__ mask1, const float* __restrict__ mask2,
                  const float* __restrict__ mask3,
                  const _Float16* __restrict__ W2h, const _Float16* __restrict__ W3h,
                  const _Float16* __restrict__ W4h,
                  _Float16* s1buf, _Float16* s2buf, _Float16* ss3h,
                  unsigned* cnt, float* out) {
  extern __shared__ _Float16 smem[];
  _Float16* W2L = smem + LDS_W2_OFF;    // this block's 128 weight rows, W2
  _Float16* W3L = smem + LDS_W3_OFF;    // this block's 128 weight rows, W3
  _Float16* AT  = smem + LDS_AT_OFF;    // per-step 16x512 spike tile

  const int lane = threadIdx.x & 31;
  const int wave = threadIdx.x >> 5;
  const int gw   = blockIdx.x * WPB + wave;                // 0..255
  const int row0 = (gw >> 5) * 16;   // batch-tile origin (8 row tiles)
  const int col0 = (gw & 31) * 16;   // hidden-tile origin (32 col tiles)
  const int n    = lane & 15;
  const int half = lane >> 4;
  const int h    = col0 + n;         // this lane's hidden unit

  // LDS-resident weights: block b needs rows h in [(b%4)*128, +128).
  const int hbase0 = (blockIdx.x & 3) * 128;
  for (int chunk = threadIdx.x; chunk < 128 * 64; chunk += TPB) {
    int row = chunk >> 6;             // 0..127
    int k   = (chunk & 63) * 8;
    *(v8h*)(W2L + row * LDS_STRIDE + k) =
        *(const v8h*)(W2h + (hbase0 + row) * HD + k);
    *(v8h*)(W3L + row * LDS_STRIDE + k) =
        *(const v8h*)(W3h + (hbase0 + row) * HD + k);
  }
  __syncthreads();
  // This wave's 16 B-matrix rows start at local row wave*16.
  const _Float16* W2w = W2L + (wave * 16) * LDS_STRIDE;
  const _Float16* W3w = W3L + (wave * 16) * LDS_STRIDE;

  const float w1v = W1[h];
  const float b1v = b1[h], b2v = b2[h], b3v = b3[h];

  // Entire recurrent state lives in WMMA C/D layout (VGPR r -> row r+8*half).
  v8f m1 = {}, m2 = {}, m3 = {};
  v8f s1p = {}, s2p = {}, s3p = {};
  v8f ss1 = {}, ss2 = {}, ss3 = {};

  unsigned target = 0;

  for (int t = 0; t < T_STEPS; ++t) {
    // ---------------- layer 1 (elementwise, W1 is [512,1]) ----------------
    {
      const float k1 = mask1[h * T_STEPS + t];
#pragma unroll
      for (int r = 0; r < 8; ++r) {
        const int brow = row0 + r + 8 * half;
        const float inp = x[brow * T_STEPS + t] * w1v + b1v;
        const float nm  = m1[r] * DECAYC * (1.0f - s1p[r]) + inp;
        const float mem = (k1 == 0.0f) ? m1[r] : nm;
        m1[r] = mem;
        const float spk = (mem > THRESH ? 1.0f : 0.0f) * k1;
        s1p[r] = spk;
        ss1[r] += spk;
        s1buf[brow * HD + h] = (_Float16)spk;
      }
    }
    target += BLOCKS; grid_sync(cnt, target);   // s1(t) visible device-wide

    // ---------------- layer 2: async-stage s1 tile, WMMA from LDS ---------
    {
      stage_tile_async(s1buf + row0 * HD, AT, threadIdx.x);
      v8f acc = gemm512(AT, LDS_STRIDE, W2w, LDS_STRIDE, lane);
      const float k2 = mask2[h * T_STEPS + t];
#pragma unroll
      for (int r = 0; r < 8; ++r) {
        const float nm  = m2[r] * DECAYC * (1.0f - s2p[r]) + acc[r] + b2v;
        const float mem = (k2 == 0.0f) ? m2[r] : nm;
        m2[r] = mem;
        const float spk = (mem > THRESH ? 1.0f : 0.0f) * k2;
        s2p[r] = spk;
        ss2[r] += spk;
        s2buf[(row0 + r + 8 * half) * HD + h] = (_Float16)spk;
      }
    }
    target += BLOCKS; grid_sync(cnt, target);   // s2(t) visible; AT reads done

    // ---------------- layer 3: async-stage s2 tile, WMMA from LDS ---------
    {
      stage_tile_async(s2buf + row0 * HD, AT, threadIdx.x);
      v8f acc = gemm512(AT, LDS_STRIDE, W3w, LDS_STRIDE, lane);
      const float k3 = mask3[h * T_STEPS + t];
#pragma unroll
      for (int r = 0; r < 8; ++r) {
        const float nm  = m3[r] * DECAYC * (1.0f - s3p[r]) + acc[r] + b3v;
        const float mem = (k3 == 0.0f) ? m3[r] : nm;
        m3[r] = mem;
        const float spk = (mem > THRESH ? 1.0f : 0.0f) * k3;
        s3p[r] = spk;
        ss3[r] += spk;
      }
    }
    // Next-step writes to s1buf/s2buf and AT reuse are ordered by the two
    // grid barriers above (each entered only after local GEMM reads finish).
  }

  // ---------------- epilogue: firing rates, layer_fr, ss3 -> f16 ----------
  const float invT = 1.0f / (float)T_STEPS;
  float* fr1 = out + NB * OUTD;
  float* fr2 = fr1 + NB * HD;
  float* fr3 = fr2 + NB * HD;
  float* layer_fr = fr3 + NB * HD;

  float p1 = 0.f, p2 = 0.f, p3 = 0.f;
#pragma unroll
  for (int r = 0; r < 8; ++r) {
    const int brow = row0 + r + 8 * half;
    fr1[brow * HD + h] = ss1[r] * invT;
    fr2[brow * HD + h] = ss2[r] * invT;
    fr3[brow * HD + h] = ss3[r] * invT;
    ss3h[brow * HD + h] = (_Float16)ss3[r];  // integer counts <= 784: f16-exact
    p1 += ss1[r]; p2 += ss2[r]; p3 += ss3[r];
  }
  const float denom = 1.0f / ((float)NB * (float)HD * (float)T_STEPS);
  atomicAdd(&layer_fr[0], p1 * denom);
  atomicAdd(&layer_fr[1], p2 * denom);
  atomicAdd(&layer_fr[2], p3 * denom);

  target += BLOCKS; grid_sync(cnt, target);   // ss3h visible

  // outputs = (Sum_t s3) @ W4^T / T + b4   (GEMM linearity; W4h zero-padded)
  if (gw < 8) {
    v8f acc = gemm512(ss3h + gw * 16 * HD, HD, W4h, HD, lane);
    if (n < OUTD) {
      const float b4v = b4[n];
#pragma unroll
      for (int r = 0; r < 8; ++r) {
        const int brow = gw * 16 + r + 8 * half;
        out[brow * OUTD + n] = acc[r] * invT + b4v;
      }
    }
  }
}

// ---------------------------------------------------------------------------
extern "C" void kernel_launch(void* const* d_in, const int* in_sizes, int n_in,
                              void* d_out, int out_size, void* d_ws, size_t ws_size,
                              hipStream_t stream) {
  const float* x     = (const float*)d_in[0];
  const float* W1    = (const float*)d_in[1];
  const float* b1    = (const float*)d_in[2];
  const float* W2    = (const float*)d_in[3];
  const float* b2    = (const float*)d_in[4];
  const float* W3    = (const float*)d_in[5];
  const float* b3    = (const float*)d_in[6];
  const float* W4    = (const float*)d_in[7];
  const float* b4    = (const float*)d_in[8];
  const float* mask1 = (const float*)d_in[9];
  const float* mask2 = (const float*)d_in[10];
  const float* mask3 = (const float*)d_in[11];

  char* ws = (char*)d_ws;
  _Float16* W2h  = (_Float16*)(ws + WS_W2H);
  _Float16* W3h  = (_Float16*)(ws + WS_W3H);
  _Float16* W4h  = (_Float16*)(ws + WS_W4H);
  _Float16* s1b  = (_Float16*)(ws + WS_S1);
  _Float16* s2b  = (_Float16*)(ws + WS_S2);
  _Float16* ss3h = (_Float16*)(ws + WS_SS3H);
  unsigned* cnt  = (unsigned*)(ws + WS_CNT);

  float* out = (float*)d_out;
  float* layer_fr = out + NB * OUTD + 3 * NB * HD;

  ffsnn_init_kernel<<<256, 256, 0, stream>>>(W2, W3, W4, W2h, W3h, W4h,
                                             cnt, layer_fr);

  ffsnn_main_kernel<<<BLOCKS, TPB, SMEM_BYTES, stream>>>(
      x, W1, b1, b2, b3, b4, mask1, mask2, mask3,
      W2h, W3h, W4h, s1b, s2b, ss3h, cnt, out);
}
